// MiniLLM_78082505441854
// MI455X (gfx1250) — matrix operand order repaired and checked by reading, not compiled, not run on hardware
//
#include <hip/hip_runtime.h>
#include <hip/hip_bf16.h>
#include <math.h>

// ---------------- model dims (compile-time) ----------------
#define BB 2
#define SS 2048
#define DD 1024
#define HH 16
#define DH 64
#define FF_ 4096
#define LL 8
#define VV 32000
#define MTOK (BB * SS)          // 4096 token rows
#define D3  (3 * DD)

typedef __attribute__((ext_vector_type(16))) _Float16 v16h;
typedef __attribute__((ext_vector_type(8)))  float    v8f;

union HF16 { v16h v; _Float16 h[16]; };
union F8   { v8f  v; float    f[8];  };
union U4H8 { uint4 u; _Float16 h[8]; };
union U2H4 { uint2 u; _Float16 h[4]; };

// =====================================================================
// Embedding: x = embed[ids] * sqrt(D) + pos ; also f16 shadow copy
// =====================================================================
__global__ __launch_bounds__(256)
void embed_kernel(const int* __restrict__ ids, const float* __restrict__ emb,
                  const float* __restrict__ pos, float* __restrict__ xo,
                  _Float16* __restrict__ xh)
{
    const int row = blockIdx.x;          // b*S + s
    const int s   = row % SS;
    const int t   = ids[row];
    const size_t rb = (size_t)row * DD;
    const int c = threadIdx.x * 4;       // 256 threads * 4 = 1024 = D
    float4 e = *(const float4*)(emb + (size_t)t * DD + c);
    float4 p = *(const float4*)(pos + (size_t)s * DD + c);
    float4 v = make_float4(e.x * 32.f + p.x, e.y * 32.f + p.y,
                           e.z * 32.f + p.z, e.w * 32.f + p.w);
    *(float4*)(xo + rb + c) = v;
    U2H4 hh; hh.h[0] = (_Float16)v.x; hh.h[1] = (_Float16)v.y;
             hh.h[2] = (_Float16)v.z; hh.h[3] = (_Float16)v.w;
    *(uint2*)(xh + rb + c) = hh.u;
}

// =====================================================================
// WMMA GEMM:  out[M,N] = act(A[M,K] @ W[N,K]^T + bias)
//   A : f16 row-major activations ; W : f32 row-major [N,K]
//   act: 0 none, 1 exact GELU ; outF (f32) / outH (f16) may be null.
// Block: 128 threads = 4 waves; tile 64x128; wave computes 32x64 (2x4 frags).
// LDS tiles stored in WMMA fragment order -> v16h frag = contiguous 32B.
// =====================================================================
#define TM 64
#define TN 128
#define TK 32

__global__ __launch_bounds__(128)
void gemm_f16_wmma(const _Float16* __restrict__ A, const float* __restrict__ W,
                   const float* __restrict__ bias, float* __restrict__ outF,
                   _Float16* __restrict__ outH, int M, int N, int K, int act)
{
    // packed layouts: [subtile][lane][slot]
    __shared__ __align__(32) _Float16 As_pk[4][32][16];   // 4 m-tiles of 16
    __shared__ __align__(32) _Float16 Bs_pk[8][32][16];   // 8 n-tiles of 16

    const int tid  = threadIdx.x;
    const int lane = tid & 31;
    const int wave = tid >> 5;
    const int wm   = wave >> 1;       // 0..1 : m offset 32*wm
    const int wn   = wave & 1;        // 0..1 : n offset 64*wn
    const int lg   = lane >> 4;
    const int ll   = lane & 15;
    const int m0   = blockIdx.y * TM;
    const int n0   = blockIdx.x * TN;

    F8 acc[2][4];
#pragma unroll
    for (int i = 0; i < 2; ++i)
#pragma unroll
        for (int j = 0; j < 4; ++j)
#pragma unroll
            for (int v = 0; v < 8; ++v) acc[i][j].f[v] = 0.0f;

    for (int k0 = 0; k0 < K; k0 += TK) {
        // ---- stage A: 64x32 halves = 256 chunks of 8 (uint4 loads) ----
#pragma unroll
        for (int it = 0; it < 2; ++it) {
            const int cid = tid + it * 128;
            const int r   = cid >> 2;
            const int c8  = (cid & 3) * 8;
            uint4 u = *(const uint4*)(A + (size_t)(m0 + r) * K + (k0 + c8));
            const int mt   = r >> 4;
            const int lned = (r & 15) + 16 * ((c8 >> 3) & 1);
            const int slot = 8 * (c8 >> 4);
            *(uint4*)&As_pk[mt][lned][slot] = u;
        }
        // ---- stage B = W^T: 32x128 f32 = 1024 chunks of 4 (float4) ----
#pragma unroll
        for (int it = 0; it < 8; ++it) {
            const int cid = tid + it * 128;
            const int n   = cid >> 3;
            const int c4  = (cid & 7) * 4;
            float4 w = *(const float4*)(W + (size_t)(n0 + n) * K + (k0 + c4));
            U2H4 hh;
            hh.h[0] = (_Float16)w.x; hh.h[1] = (_Float16)w.y;
            hh.h[2] = (_Float16)w.z; hh.h[3] = (_Float16)w.w;
            const int nt   = n >> 4;
            const int lned = (n & 15) + 16 * ((c4 >> 3) & 1);
            const int slot = (c4 & 7) + 8 * (c4 >> 4);
            *(uint2*)&Bs_pk[nt][lned][slot] = hh.u;
        }
        __syncthreads();

        // ---- fragments: contiguous 32B per lane ----
        HF16 a[2], b[4];
#pragma unroll
        for (int i = 0; i < 2; ++i)
            a[i].v = *(const v16h*)&As_pk[wm * 2 + i][lane][0];
#pragma unroll
        for (int j = 0; j < 4; ++j)
            b[j].v = *(const v16h*)&Bs_pk[wn * 4 + j][lane][0];
#pragma unroll
        for (int i = 0; i < 2; ++i)
#pragma unroll
            for (int j = 0; j < 4; ++j)
                acc[i][j].v = __builtin_amdgcn_wmma_f32_16x16x32_f16(
                    false, a[i].v, false, b[j].v, (short)0, acc[i][j].v,
                    false, false);
        __syncthreads();
    }

    // ---- epilogue ----
#pragma unroll
    for (int i = 0; i < 2; ++i)
#pragma unroll
        for (int j = 0; j < 4; ++j) {
            const int gn = n0 + wn * 64 + j * 16 + ll;
            const float bb = bias ? bias[gn] : 0.0f;
#pragma unroll
            for (int v = 0; v < 8; ++v) {
                const int gm = m0 + wm * 32 + i * 16 + v + (lg << 3);
                float x = acc[i][j].f[v] + bb;
                if (act == 1)
                    x = 0.5f * x * (1.0f + erff(x * 0.70710678118654752f));
                const size_t idx = (size_t)gm * N + gn;
                if (outF) outF[idx] = x;
                if (outH) outH[idx] = (_Float16)x;
            }
        }
}

// =====================================================================
// Flash attention (causal): block per (b, h, 64-row q tile); 4 waves.
// qkv: [B,S,3D] f16. LDS K/V/P tiles kept in fragment-packed order.
// =====================================================================
__global__ __launch_bounds__(128)
void attn_kernel(const _Float16* __restrict__ qkv, _Float16* __restrict__ o)
{
    // [tile][kstep][lane][slot]
    __shared__ __align__(32) _Float16 Ks_pk[4][2][32][16];
    __shared__ __align__(32) _Float16 Vs_pk[4][2][32][16];
    __shared__ __align__(32) _Float16 Ps_pk[4][2][32][16];

    const int tid  = threadIdx.x;
    const int lane = tid & 31;
    const int wave = tid >> 5;
    const int lg   = lane >> 4;
    const int ll   = lane & 15;
    const int qb   = blockIdx.x * 64;
    const int b    = blockIdx.y / HH;
    const int h    = blockIdx.y % HH;
    const float scale = 0.125f;   // 1/sqrt(DH)

    // ---- Q fragments (A layout) straight from global, 16B loads ----
    HF16 aq[2];
    {
        const int s = qb + wave * 16 + ll;
        const _Float16* qp = qkv + (size_t)(b * SS + s) * D3 + h * DH;
        const int kb = lg * 8;
#pragma unroll
        for (int ks = 0; ks < 2; ++ks) {
            U4H8 lo, hi;
            lo.u = *(const uint4*)(qp + ks * 32 + kb);
            hi.u = *(const uint4*)(qp + ks * 32 + kb + 16);
#pragma unroll
            for (int e = 0; e < 8; ++e) {
                aq[ks].h[e]     = lo.h[e];
                aq[ks].h[8 + e] = hi.h[e];
            }
        }
    }

    F8 oacc[4];
    float rmax[8], rsum[8];
#pragma unroll
    for (int d = 0; d < 4; ++d)
#pragma unroll
        for (int v = 0; v < 8; ++v) oacc[d].f[v] = 0.0f;
#pragma unroll
    for (int v = 0; v < 8; ++v) { rmax[v] = -1e30f; rsum[v] = 0.0f; }

    const int nkt = blockIdx.x + 1;
    for (int kt = 0; kt < nkt; ++kt) {
        __syncthreads();
        const int s0 = kt * 64;
        // ---- stage K (B-layout for QK^T): 8-half chunks along dh ----
#pragma unroll
        for (int it = 0; it < 4; ++it) {
            const int cid = tid + it * 128;
            const int r   = cid >> 3;          // kv row
            const int c8  = (cid & 7) * 8;     // dh base
            uint4 u = *(const uint4*)(qkv + (size_t)(b * SS + s0 + r) * D3
                                      + DD + h * DH + c8);
            const int nt   = r >> 4;
            const int ks   = c8 >> 5;
            const int lned = (r & 15) + 16 * ((c8 >> 3) & 1);
            const int slot = 8 * ((c8 >> 4) & 1);
            *(uint4*)&Ks_pk[nt][ks][lned][slot] = u;
        }
        // ---- stage V (B-layout for P·V): 8-kv-row runs, fixed dh col ----
#pragma unroll
        for (int it = 0; it < 4; ++it) {
            const int cid = tid + it * 128;
            const int c   = cid & 63;          // dh col
            const int r8  = (cid >> 6) * 8;    // kv base
            U4H8 pk;
#pragma unroll
            for (int e = 0; e < 8; ++e)
                pk.h[e] = qkv[(size_t)(b * SS + s0 + r8 + e) * D3
                              + 2 * DD + h * DH + c];
            const int nt   = c >> 4;
            const int ks   = r8 >> 5;
            const int lned = (c & 15) + 16 * ((r8 >> 3) & 1);
            const int slot = 8 * ((r8 >> 4) & 1);
            *(uint4*)&Vs_pk[nt][ks][lned][slot] = pk.u;
        }
        __syncthreads();

        // ---- scores S = Q K^T ----
        F8 sfr[4];
#pragma unroll
        for (int nt = 0; nt < 4; ++nt) {
#pragma unroll
            for (int v = 0; v < 8; ++v) sfr[nt].f[v] = 0.0f;
#pragma unroll
            for (int ks = 0; ks < 2; ++ks) {
                HF16 bf;
                bf.v = *(const v16h*)&Ks_pk[nt][ks][lane][0];
                sfr[nt].v = __builtin_amdgcn_wmma_f32_16x16x32_f16(
                    false, aq[ks].v, false, bf.v, (short)0, sfr[nt].v,
                    false, false);
            }
#pragma unroll
            for (int v = 0; v < 8; ++v) sfr[nt].f[v] *= scale;
        }

        // ---- causal mask (diagonal tile only) ----
        if (kt == blockIdx.x) {
#pragma unroll
            for (int nt = 0; nt < 4; ++nt)
#pragma unroll
                for (int v = 0; v < 8; ++v) {
                    const int grow = qb + wave * 16 + v + lg * 8;
                    const int gcol = s0 + nt * 16 + ll;
                    if (gcol > grow) sfr[nt].f[v] = -1e30f;
                }
        }

        // ---- online softmax (row = v + 8*lg, 16 lanes per group) ----
#pragma unroll
        for (int v = 0; v < 8; ++v) {
            float m = fmaxf(fmaxf(sfr[0].f[v], sfr[1].f[v]),
                            fmaxf(sfr[2].f[v], sfr[3].f[v]));
#pragma unroll
            for (int off = 1; off < 16; off <<= 1)
                m = fmaxf(m, __shfl_xor(m, off, 32));
            const float mnew  = fmaxf(rmax[v], m);
            const float alpha = __expf(rmax[v] - mnew);
            rmax[v] = mnew;
            float psum = 0.0f;
#pragma unroll
            for (int nt = 0; nt < 4; ++nt) {
                const float p = __expf(sfr[nt].f[v] - mnew);
                sfr[nt].f[v] = p;
                psum += p;
            }
#pragma unroll
            for (int off = 1; off < 16; off <<= 1)
                psum += __shfl_xor(psum, off, 32);
            rsum[v] = rsum[v] * alpha + psum;
#pragma unroll
            for (int d = 0; d < 4; ++d) oacc[d].f[v] *= alpha;
        }

        // ---- relayout P: C-layout f32 -> A-layout f16 (wave-private) ----
#pragma unroll
        for (int nt = 0; nt < 4; ++nt)
#pragma unroll
            for (int v = 0; v < 8; ++v) {
                const int m    = v + 8 * lg;
                const int ksd  = nt >> 1;
                const int lned = m + 16 * ((ll >> 3) & 1);
                const int slot = (ll & 7) + 8 * (nt & 1);
                Ps_pk[wave][ksd][lned][slot] = (_Float16)sfr[nt].f[v];
            }

        // ---- O += P V ----
#pragma unroll
        for (int d = 0; d < 4; ++d) {
#pragma unroll
            for (int ks = 0; ks < 2; ++ks) {
                HF16 af, bf;
                af.v = *(const v16h*)&Ps_pk[wave][ks][lane][0];
                bf.v = *(const v16h*)&Vs_pk[d][ks][lane][0];
                oacc[d].v = __builtin_amdgcn_wmma_f32_16x16x32_f16(
                    false, af.v, false, bf.v, (short)0, oacc[d].v,
                    false, false);
            }
        }
    }

    // ---- normalize and write O (f16, head-concat columns) ----
#pragma unroll
    for (int d = 0; d < 4; ++d)
#pragma unroll
        for (int v = 0; v < 8; ++v) {
            const int gs = qb + wave * 16 + v + lg * 8;
            const int gc = h * DH + d * 16 + ll;
            o[(size_t)(b * SS + gs) * DD + gc] =
                (_Float16)(oacc[d].f[v] / rsum[v]);
        }
}

// =====================================================================
// x = LayerNorm(res + add) * w + b ; write f32 + f16 shadow
// =====================================================================
__global__ __launch_bounds__(256)
void add_ln_kernel(const float* __restrict__ res, const float* __restrict__ add,
                   const float* __restrict__ w, const float* __restrict__ bvec,
                   float* __restrict__ xo, _Float16* __restrict__ xh)
{
    __shared__ float red[256];
    const int row = blockIdx.x;
    const int tid = threadIdx.x;
    const size_t rb = (size_t)row * DD;
    const int c = tid * 4;

    float4 r4 = *(const float4*)(res + rb + c);
    float4 a4 = *(const float4*)(add + rb + c);
    float vals[4] = { r4.x + a4.x, r4.y + a4.y, r4.z + a4.z, r4.w + a4.w };
    float s = vals[0] + vals[1] + vals[2] + vals[3];
    red[tid] = s;
    __syncthreads();
    for (int off = 128; off > 0; off >>= 1) {
        if (tid < off) red[tid] += red[tid + off];
        __syncthreads();
    }
    const float mu = red[0] * (1.0f / DD);
    __syncthreads();

    float s2 = 0.0f;
#pragma unroll
    for (int i = 0; i < 4; ++i) { const float d = vals[i] - mu; s2 += d * d; }
    red[tid] = s2;
    __syncthreads();
    for (int off = 128; off > 0; off >>= 1) {
        if (tid < off) red[tid] += red[tid + off];
        __syncthreads();
    }
    const float rs = rsqrtf(red[0] * (1.0f / DD) + 1e-5f);

    float4 w4 = *(const float4*)(w + c);
    float4 b4 = *(const float4*)(bvec + c);
    float4 y = make_float4((vals[0] - mu) * rs * w4.x + b4.x,
                           (vals[1] - mu) * rs * w4.y + b4.y,
                           (vals[2] - mu) * rs * w4.z + b4.z,
                           (vals[3] - mu) * rs * w4.w + b4.w);
    *(float4*)(xo + rb + c) = y;
    U2H4 hh; hh.h[0] = (_Float16)y.x; hh.h[1] = (_Float16)y.y;
             hh.h[2] = (_Float16)y.z; hh.h[3] = (_Float16)y.w;
    *(uint2*)(xh + rb + c) = hh.u;
}

// =====================================================================
// host-side orchestration
// =====================================================================
extern "C" void kernel_launch(void* const* d_in, const int* in_sizes, int n_in,
                              void* d_out, int out_size, void* d_ws, size_t ws_size,
                              hipStream_t stream)
{
    (void)in_sizes; (void)n_in; (void)out_size; (void)ws_size;

    const int*   input_ids  = (const int*)  d_in[0];
    const float* embed_tab  = (const float*)d_in[1];
    const float* pos_embed  = (const float*)d_in[2];
    const float* in_proj_w  = (const float*)d_in[3];
    const float* in_proj_b  = (const float*)d_in[4];
    const float* out_proj_w = (const float*)d_in[5];
    const float* out_proj_b = (const float*)d_in[6];
    const float* ln1_w      = (const float*)d_in[7];
    const float* ln1_b      = (const float*)d_in[8];
    const float* ff1_w      = (const float*)d_in[9];
    const float* ff1_b      = (const float*)d_in[10];
    const float* ff2_w      = (const float*)d_in[11];
    const float* ff2_b      = (const float*)d_in[12];
    const float* ln2_w      = (const float*)d_in[13];
    const float* ln2_b      = (const float*)d_in[14];
    const float* lm_head_w  = (const float*)d_in[15];

    char* ws = (char*)d_ws;
    size_t off = 0;
    float*     xf   = (float*)(ws + off);     off += (size_t)MTOK * DD * 4;
    _Float16*  xh   = (_Float16*)(ws + off);  off += (size_t)MTOK * DD * 2;
    _Float16*  qkvh = (_Float16*)(ws + off);  off += (size_t)MTOK * D3 * 2;
    _Float16*  oh   = (_Float16*)(ws + off);  off += (size_t)MTOK * DD * 2;
    float*     tmp  = (float*)(ws + off);     off += (size_t)MTOK * DD * 4;
    _Float16*  ffh  = (_Float16*)(ws + off);  off += (size_t)MTOK * FF_ * 2;

    const dim3 gQKV(D3 / TN,  MTOK / TM);   // (24, 64)
    const dim3 gPRJ(DD / TN,  MTOK / TM);   // (8, 64)
    const dim3 gFF1(FF_ / TN, MTOK / TM);   // (32, 64)
    const dim3 gLMH(VV / TN,  MTOK / TM);   // (250, 64)
    const dim3 gATT(SS / 64, BB * HH);      // (32, 32)

    embed_kernel<<<MTOK, 256, 0, stream>>>(input_ids, embed_tab, pos_embed, xf, xh);

    for (int l = 0; l < LL; ++l) {
        gemm_f16_wmma<<<gQKV, 128, 0, stream>>>(
            xh, in_proj_w + (size_t)l * D3 * DD, in_proj_b + (size_t)l * D3,
            nullptr, qkvh, MTOK, D3, DD, 0);
        attn_kernel<<<gATT, 128, 0, stream>>>(qkvh, oh);
        gemm_f16_wmma<<<gPRJ, 128, 0, stream>>>(
            oh, out_proj_w + (size_t)l * DD * DD, out_proj_b + (size_t)l * DD,
            tmp, nullptr, MTOK, DD, DD, 0);
        add_ln_kernel<<<MTOK, 256, 0, stream>>>(
            xf, tmp, ln1_w + (size_t)l * DD, ln1_b + (size_t)l * DD, xf, xh);
        gemm_f16_wmma<<<gFF1, 128, 0, stream>>>(
            xh, ff1_w + (size_t)l * FF_ * DD, ff1_b + (size_t)l * FF_,
            nullptr, ffh, MTOK, FF_, DD, 1);
        gemm_f16_wmma<<<gPRJ, 128, 0, stream>>>(
            ffh, ff2_w + (size_t)l * DD * FF_, ff2_b + (size_t)l * DD,
            tmp, nullptr, MTOK, DD, FF_, 0);
        add_ln_kernel<<<MTOK, 256, 0, stream>>>(
            xf, tmp, ln2_w + (size_t)l * DD, ln2_b + (size_t)l * DD, xf, xh);
    }

    gemm_f16_wmma<<<gLMH, 128, 0, stream>>>(
        xh, lm_head_w, nullptr, (float*)d_out, nullptr, MTOK, VV, DD, 0);
}
